// MultiHeadAttention_60790967108212
// MI455X (gfx1250) — matrix operand order repaired and checked by reading, not compiled, not run on hardware
//
#include <hip/hip_runtime.h>
#include <hip/hip_bf16.h>

typedef _Float16 v16h __attribute__((ext_vector_type(16)));
typedef _Float16 v2h  __attribute__((ext_vector_type(2)));
typedef float    v8f  __attribute__((ext_vector_type(8)));

#define NHEADS 8
#define DHEAD  32
#define NPIX   4096
#define CDIM   256

// K-offset of VGPR pair v for lane-group g in the contiguous k order (k = 16g + 2v) this file uses for A and B
__device__ __forceinline__ int a_kbase(int v, int g) {
    return 16 * g + 2 * v;
}

__device__ __forceinline__ v8f wmma_f16(const v16h& a, const v16h& b, const v8f& c) {
    return __builtin_amdgcn_wmma_f32_16x16x32_f16(false, a, false, b, (short)0, c, false, false);
}

// ---------------------------------------------------------------------------
// Kernel 0: fp32 weights -> f16, pre-swizzled into per-lane A-tile order.
// Output index e = (((T*8 + kb)*16 + m)*2 + g)*16 + halfidx ; source W[T*16+m][kb*32+K]
// ---------------------------------------------------------------------------
__global__ __launch_bounds__(256) void swizzle_w_kernel(const float* __restrict__ w,
                                                        _Float16* __restrict__ wa,
                                                        int mtiles) {
    int e = blockIdx.x * blockDim.x + threadIdx.x;
    int total = mtiles * 16 * 256;
    if (e >= total) return;
    int hidx = e & 15;
    int g    = (e >> 4) & 1;
    int m    = (e >> 5) & 15;
    int kb   = (e >> 9) & 7;
    int T    = e >> 12;
    int v = hidx >> 1, j = hidx & 1;
    int K = a_kbase(v, g) + j;
    wa[e] = (_Float16)w[(T * 16 + m) * 256 + kb * 32 + K];
}

// ---------------------------------------------------------------------------
// Kernel 1: QKV projection. Grid = B * (N/16) blocks of 256 threads (8 waves).
// Each block: 16 pixels, all 768 output channels (8 waves x 6 tiles).
// Writes q (scaled by log2(e)/sqrt(d)), k row-major [bh][n][d],
// v transposed [bh][d][n] as f16.
// ---------------------------------------------------------------------------
__global__ __launch_bounds__(256) void qkv_proj_kernel(const float* __restrict__ x,
                                                       const _Float16* __restrict__ wa,
                                                       const float* __restrict__ bias,
                                                       _Float16* __restrict__ qbuf,
                                                       _Float16* __restrict__ kbuf,
                                                       _Float16* __restrict__ vtbuf) {
    __shared__ _Float16 lx[512];  // [pixel p][channel cc] as lx[p*32 + cc]
    const int b    = blockIdx.x >> 8;
    const int n0   = (blockIdx.x & 255) * 16;
    const int t    = threadIdx.x;
    const int lane = t & 31;
    const int wv   = t >> 5;
    const int m16  = lane & 15;
    const int g    = lane >> 4;

    v8f zero = {};
    v8f acc[6];
#pragma unroll
    for (int i = 0; i < 6; ++i) acc[i] = zero;

    for (int kb = 0; kb < 8; ++kb) {
        {   // cooperative stage of x block (32 ch x 16 px) into LDS, f32 -> f16
            int cc = t >> 3;
            int pp = (t & 7) * 2;
            const float2 xv =
                *(const float2*)&x[((size_t)(b * CDIM + kb * 32 + cc)) * NPIX + n0 + pp];
            lx[pp * 32 + cc]       = (_Float16)xv.x;
            lx[(pp + 1) * 32 + cc] = (_Float16)xv.y;
        }
        __syncthreads();
        // B tile: lane = pixel m16, contiguous K halves for its group
        const v16h bt = *(const v16h*)&lx[m16 * 32 + g * 16];
#pragma unroll
        for (int tt = 0; tt < 6; ++tt) {
            int T = wv * 6 + tt;
            const v16h at = *(const v16h*)&wa[(((T * 8 + kb) * 16 + m16) * 2 + g) * 16];
            acc[tt] = wmma_f16(at, bt, acc[tt]);
        }
        __syncthreads();
    }

    // 1/sqrt(32) * log2(e): softmax done in log2 domain -> bare v_exp_f32
    const float qscale = 0.25500052395775595f;
    const int n = n0 + m16;
#pragma unroll
    for (int tt = 0; tt < 6; ++tt) {
        int T = wv * 6 + tt;
#pragma unroll
        for (int r = 0; r < 8; ++r) {
            int o = T * 16 + r + 8 * g;
            float val = acc[tt][r] + bias[o];
            if (o < 256) {
                int h = o >> 5, dd = o & 31;
                qbuf[(((size_t)(b * NHEADS + h)) * NPIX + n) * DHEAD + dd] =
                    (_Float16)(val * qscale);
            } else if (o < 512) {
                int c = o - 256, h = c >> 5, dd = c & 31;
                kbuf[(((size_t)(b * NHEADS + h)) * NPIX + n) * DHEAD + dd] = (_Float16)val;
            } else {
                int c = o - 512, h = c >> 5, dd = c & 31;
                vtbuf[(((size_t)(b * NHEADS + h)) * DHEAD + dd) * NPIX + n] = (_Float16)val;
            }
        }
    }
}

// ---------------------------------------------------------------------------
// Kernel 2: flash attention, 64 keys per softmax round (8 WMMAs / round).
// Grid = (B*h) * (N/16) single-wave blocks. Softmax in log2 domain.
// P bounced through LDS (16x64 f16) into A-layout.
// Output written in the torch-faithful scrambled (B,C,HW) layout for proj.
// ---------------------------------------------------------------------------
__global__ __launch_bounds__(32) void flash_attn_kernel(const _Float16* __restrict__ qbuf,
                                                        const _Float16* __restrict__ kbuf,
                                                        const _Float16* __restrict__ vtbuf,
                                                        _Float16* __restrict__ pin) {
    __shared__ _Float16 lp[16 * 64];  // P tile [m][kk], kk in [0,64)
    const int bh   = blockIdx.x >> 8;
    const int i0   = (blockIdx.x & 255) * 16;
    const int lane = threadIdx.x;
    const int m16  = lane & 15;
    const int g    = lane >> 4;

    const _Float16* kbase = &kbuf[((size_t)bh) * NPIX * DHEAD];
    const _Float16* vbase = &vtbuf[((size_t)bh) * DHEAD * NPIX];

    // Q A-tile (loaded once)
    v16h aq;
    {
        const _Float16* qrow = &qbuf[(((size_t)bh) * NPIX + i0 + m16) * DHEAD];
#pragma unroll
        for (int v = 0; v < 8; ++v) {
            v2h tv = *(const v2h*)&qrow[a_kbase(v, g)];
            aq[2 * v]     = tv[0];
            aq[2 * v + 1] = tv[1];
        }
    }

    float mi[8], li[8];
#pragma unroll
    for (int r = 0; r < 8; ++r) { mi[r] = -1e30f; li[r] = 0.0f; }
    v8f acc0 = {}, acc1 = {};
    v8f zero = {};

    for (int j0 = 0; j0 < NPIX; j0 += 64) {
        // ---- S = Q K^T over 64 keys: 4 tiles ----
        v8f s[4];
        {
            const size_t krow = (size_t)(j0 + m16) * DHEAD + g * 16;
#pragma unroll
            for (int tk = 0; tk < 4; ++tk) {
                const v16h bk = *(const v16h*)&kbase[krow + (size_t)tk * (16 * DHEAD)];
                s[tk] = wmma_f16(aq, bk, zero);
            }
        }

        // ---- online softmax (log2 domain) ----
#pragma unroll
        for (int r = 0; r < 8; ++r) {
            float mx = fmaxf(fmaxf(s[0][r], s[1][r]), fmaxf(s[2][r], s[3][r]));
            mx = fmaxf(mx, __shfl_xor(mx, 1));
            mx = fmaxf(mx, __shfl_xor(mx, 2));
            mx = fmaxf(mx, __shfl_xor(mx, 4));
            mx = fmaxf(mx, __shfl_xor(mx, 8));
            float mnew  = fmaxf(mi[r], mx);
            float alpha = exp2f(mi[r] - mnew);
            float p0 = exp2f(s[0][r] - mnew);
            float p1 = exp2f(s[1][r] - mnew);
            float p2 = exp2f(s[2][r] - mnew);
            float p3 = exp2f(s[3][r] - mnew);
            float rs = (p0 + p1) + (p2 + p3);
            rs += __shfl_xor(rs, 1);
            rs += __shfl_xor(rs, 2);
            rs += __shfl_xor(rs, 4);
            rs += __shfl_xor(rs, 8);
            li[r] = li[r] * alpha + rs;
            mi[r] = mnew;
            acc0[r] *= alpha;
            acc1[r] *= alpha;
            int prow = (r + 8 * g) * 64 + m16;
            lp[prow]      = (_Float16)p0;
            lp[prow + 16] = (_Float16)p1;
            lp[prow + 32] = (_Float16)p2;
            lp[prow + 48] = (_Float16)p3;
        }
        __syncthreads();

        // ---- O += P V : two P A-tiles (kk 0..31, 32..63), V B-tiles ----
        v16h ap0, ap1;
#pragma unroll
        for (int v = 0; v < 8; ++v) {
            int kb = a_kbase(v, g);
            v2h t0 = *(const v2h*)&lp[m16 * 64 + kb];
            v2h t1 = *(const v2h*)&lp[m16 * 64 + 32 + kb];
            ap0[2 * v] = t0[0]; ap0[2 * v + 1] = t0[1];
            ap1[2 * v] = t1[0]; ap1[2 * v + 1] = t1[1];
        }
        {
            const size_t v00 = (size_t)m16 * NPIX + j0 + g * 16;
            const size_t v01 = (size_t)(16 + m16) * NPIX + j0 + g * 16;
            const v16h bv00 = *(const v16h*)&vbase[v00];
            const v16h bv01 = *(const v16h*)&vbase[v01];
            acc0 = wmma_f16(ap0, bv00, acc0);
            acc1 = wmma_f16(ap0, bv01, acc1);
            const v16h bv10 = *(const v16h*)&vbase[v00 + 32];
            const v16h bv11 = *(const v16h*)&vbase[v01 + 32];
            acc0 = wmma_f16(ap1, bv10, acc0);
            acc1 = wmma_f16(ap1, bv11, acc1);
        }
        __syncthreads();
    }

    const int b = bh >> 3, h = bh & 7;
#pragma unroll
    for (int r = 0; r < 8; ++r) {
        float inv = 1.0f / li[r];
        int i = i0 + r + 8 * g;
        // torch-faithful: flat = n*256 + h*32 + dd -> (c', n') = (flat>>12, flat&4095)
        size_t base = ((size_t)(b * CDIM + (i >> 4))) * NPIX + (i & 15) * 256 + h * 32;
        pin[base + m16]      = (_Float16)(acc0[r] * inv);
        pin[base + 16 + m16] = (_Float16)(acc1[r] * inv);
    }
}

// ---------------------------------------------------------------------------
// Kernel 3: output projection (256x256x8192 GEMM) -> fp32 d_out + bias.
// ---------------------------------------------------------------------------
__global__ __launch_bounds__(256) void proj_kernel(const _Float16* __restrict__ pin,
                                                   const _Float16* __restrict__ wa,
                                                   const float* __restrict__ bias,
                                                   float* __restrict__ out) {
    __shared__ _Float16 lx[512];
    const int b    = blockIdx.x >> 8;
    const int n0   = (blockIdx.x & 255) * 16;
    const int t    = threadIdx.x;
    const int lane = t & 31;
    const int wv   = t >> 5;
    const int m16  = lane & 15;
    const int g    = lane >> 4;

    v8f acc0 = {}, acc1 = {};
    for (int kb = 0; kb < 8; ++kb) {
        {
            int cc = t >> 3;
            int pp = (t & 7) * 2;
            v2h pv = *(const v2h*)&pin[((size_t)(b * CDIM + kb * 32 + cc)) * NPIX + n0 + pp];
            lx[pp * 32 + cc]       = pv[0];
            lx[(pp + 1) * 32 + cc] = pv[1];
        }
        __syncthreads();
        const v16h bt = *(const v16h*)&lx[m16 * 32 + g * 16];
        {
            int T0 = wv * 2, T1 = wv * 2 + 1;
            const v16h at0 = *(const v16h*)&wa[(((T0 * 8 + kb) * 16 + m16) * 2 + g) * 16];
            acc0 = wmma_f16(at0, bt, acc0);
            const v16h at1 = *(const v16h*)&wa[(((T1 * 8 + kb) * 16 + m16) * 2 + g) * 16];
            acc1 = wmma_f16(at1, bt, acc1);
        }
        __syncthreads();
    }

    const int n = n0 + m16;
#pragma unroll
    for (int r = 0; r < 8; ++r) {
        int o0 = (wv * 2) * 16 + r + 8 * g;
        out[((size_t)(b * CDIM + o0)) * NPIX + n] = acc0[r] + bias[o0];
        int o1 = (wv * 2 + 1) * 16 + r + 8 * g;
        out[((size_t)(b * CDIM + o1)) * NPIX + n] = acc1[r] + bias[o1];
    }
}

// ---------------------------------------------------------------------------
extern "C" void kernel_launch(void* const* d_in, const int* in_sizes, int n_in,
                              void* d_out, int out_size, void* d_ws, size_t ws_size,
                              hipStream_t stream) {
    const float* x      = (const float*)d_in[0];
    const float* w_qkv  = (const float*)d_in[1];
    const float* b_qkv  = (const float*)d_in[2];
    const float* w_proj = (const float*)d_in[3];
    const float* b_proj = (const float*)d_in[4];
    float* out = (float*)d_out;

    char* ws = (char*)d_ws;
    _Float16* wa_qkv  = (_Float16*)(ws);                               // 768*256*2 = 384 KB
    _Float16* wa_proj = (_Float16*)(ws + 393216);                      // 256*256*2 = 128 KB
    _Float16* q_buf   = (_Float16*)(ws + 524288);                      // 4 MB
    _Float16* k_buf   = (_Float16*)(ws + 524288 + 1 * 4194304);        // 4 MB
    _Float16* vt_buf  = (_Float16*)(ws + 524288 + 2 * 4194304);        // 4 MB
    _Float16* pin     = (_Float16*)(ws + 524288 + 3 * 4194304);        // 4 MB

    swizzle_w_kernel<<<dim3(768), dim3(256), 0, stream>>>(w_qkv, wa_qkv, 48);
    swizzle_w_kernel<<<dim3(256), dim3(256), 0, stream>>>(w_proj, wa_proj, 16);
    qkv_proj_kernel<<<dim3(512), dim3(256), 0, stream>>>(x, wa_qkv, b_qkv,
                                                         q_buf, k_buf, vt_buf);
    flash_attn_kernel<<<dim3(4096), dim3(32), 0, stream>>>(q_buf, k_buf, vt_buf, pin);
    proj_kernel<<<dim3(512), dim3(256), 0, stream>>>(pin, wa_proj, b_proj, out);
}
